// MsAutoformer_EncoderOnly_66846870995207
// MI455X (gfx1250) — matrix operand order repaired and checked by reading, not compiled, not run on hardware
//
#include <hip/hip_runtime.h>
#include <hip/hip_bf16.h>
#include <math.h>

typedef _Float16 half_t;
typedef __attribute__((ext_vector_type(16))) _Float16 v16h;
typedef __attribute__((ext_vector_type(8)))  _Float16 v8h;
typedef __attribute__((ext_vector_type(8)))  float    v8f;

#define DEV __device__ __forceinline__

constexpr int BB = 64, LL = 1024, CIN = 16, DM = 256, DFF = 1024, KTOP = 20;
constexpr int MROWS = BB * LL; // 65536

// ---------------- workspace layout (bytes) ----------------
constexpr size_t MB = 1ull << 20;
constexpr size_t OFF_ENC32 = 0;        // 64MB  f32 [65536,256]; aliased as XS32
constexpr size_t OFF_ENC16 = 64*MB;    // 32MB  f16 [65536,256]; aliased as XS16
constexpr size_t OFF_Q16   = 96*MB;    // 32MB  f16; reused as AC16
constexpr size_t OFF_K16   = 128*MB;   // 32MB
constexpr size_t OFF_V16   = 160*MB;   // 32MB
constexpr size_t OFF_X32   = 192*MB;   // 64MB  f32 scratch
constexpr size_t OFF_Y16   = 256*MB;   // 128MB f16 FFN hidden; also conv1 output H1
constexpr size_t OFF_WP    = 384*MB;   // ~15MB f16 packed weights
constexpr size_t OFF_SM    = 400*MB;   // small f32/int pool
// weight pool sub-offsets (in halves)
constexpr size_t WP_WQ = 0, WP_WK = 131072, WP_WV = 262144, WP_WO = 393216;
constexpr size_t WP_FF1 = 524288, WP_FF2 = 1048576, WP_C2 = 1572864;
constexpr size_t WP_PROJ = 1769472, WP_RP1 = 5963776, WP_XE = 6160384, WP_CAT = 7208960;
// small pool sub-offsets (bytes)
constexpr size_t SM_MEANV = 0, SM_W = 262144, SM_D = 267264, SM_P1 = 272384;
constexpr size_t SM_P2 = 337920, SM_EPRE = 403456, SM_EMB = 468992;
constexpr size_t SM_RPRE = 534528, SM_RPN = 600064;

// ---------------- WMMA fragment helpers (wave32; k contiguous per lane half, same order in A and B) ----------------
// A 16x32 f16: lanes 0-15 M=lane K={0..15}; lanes 16-31 M=lane-16 K={16..31}
DEV v16h load_a(const half_t* A, int lda, int row, int k0, int lane) {
  const half_t* p = A + (size_t)row * lda + k0 + ((lane & 16) ? 16 : 0);
  union { v16h v; v8h h[2]; } f;
  f.h[0] = *(const v8h*)p;
  f.h[1] = *(const v8h*)(p + 8);
  return f.v;
}
// B 32x16 f16 from Bt[N,K] row-major: lane n=n0+(lane&15); halves i -> K = k0+(lane&16?16:0)+i
DEV v16h load_b(const half_t* Bt, int ldb, int n0, int k0, int lane) {
  const half_t* p = Bt + (size_t)(n0 + (lane & 15)) * ldb + k0 + ((lane & 16) ? 16 : 0);
  return *(const v16h*)p;
}
DEV v8f wmma16(v16h a, v16h b, v8f c) {
  return __builtin_amdgcn_wmma_f32_16x16x32_f16(false, a, false, b, (short)0, c, false, false);
}

// ---------------- weight pack: dst[n*K+k] = (f16) src[n*sn + k*sk + off] ----------------
__global__ void k_pack(const float* __restrict__ src, half_t* __restrict__ dst,
                       int N, int K, int sn, int sk, int off) {
  int g = blockIdx.x * blockDim.x + threadIdx.x;
  if (g >= N * K) return;
  int n = g / K, k = g % K;
  dst[g] = (half_t)src[(size_t)n * sn + (size_t)k * sk + off];
}

__global__ void k_zero(float* p, int n) {
  int i = blockIdx.x * blockDim.x + threadIdx.x;
  if (i < n) p[i] = 0.f;
}

// ------- generic WMMA GEMM, 2x2 register tiling (32x32 per wave) -------
// out = act(A[M,K] @ Bt[N,K]^T + bias + resid)
__global__ void k_gemm(const half_t* __restrict__ A, const half_t* __restrict__ Bt,
                       const float* __restrict__ bias, const float* __restrict__ resid,
                       float* __restrict__ out32, half_t* __restrict__ out16,
                       int M, int N, int Kd, int act) {
  int wid = (blockIdx.x * blockDim.x + threadIdx.x) >> 5;
  int lane = threadIdx.x & 31;
  int ntn = N >> 5;
  if (wid >= (M >> 5) * ntn) return;
  int row0 = (wid / ntn) << 5, n0 = (wid % ntn) << 5;
  v8f c00 = {}, c01 = {}, c10 = {}, c11 = {};
#pragma unroll 2
  for (int k0 = 0; k0 < Kd; k0 += 32) {
    v16h a0 = load_a(A, Kd, row0 + (lane & 15), k0, lane);
    v16h a1 = load_a(A, Kd, row0 + 16 + (lane & 15), k0, lane);
    v16h b0 = load_b(Bt, Kd, n0, k0, lane);
    v16h b1 = load_b(Bt, Kd, n0 + 16, k0, lane);
    c00 = wmma16(a0, b0, c00);
    c01 = wmma16(a0, b1, c01);
    c10 = wmma16(a1, b0, c10);
    c11 = wmma16(a1, b1, c11);
  }
  v8f cc[4] = {c00, c01, c10, c11};
  int nl = lane & 15, mb = (lane & 16) ? 8 : 0;
#pragma unroll
  for (int tt = 0; tt < 4; ++tt) {
    int rbase = row0 + ((tt >> 1) << 4) + mb;
    int n = n0 + ((tt & 1) << 4) + nl;
    float bv = bias ? bias[n] : 0.f;
#pragma unroll
    for (int r = 0; r < 8; ++r) {
      int m = rbase + r;
      float v = cc[tt][r] + bv;
      if (resid) v += resid[(size_t)m * N + n];
      if (act == 1) v = fmaxf(v, 0.f);
      else if (act == 2) v = 0.5f * v * (1.f + erff(v * 0.70710678118f)); // exact gelu
      if (out32) out32[(size_t)m * N + n] = v;
      if (out16) out16[(size_t)m * N + n] = (half_t)v;
    }
  }
}

// ---------------- conv1: 16->256, k=3, zero pad, relu; direct VALU ----------------
__global__ void k_conv1(const float* __restrict__ x, const float* __restrict__ w,
                        const float* __restrict__ bias, half_t* __restrict__ h1) {
  int gid = blockIdx.x * blockDim.x + threadIdx.x;
  if (gid >= MROWS * DM) return;
  int co = gid & 255, l = (gid >> 8) & 1023, b = gid >> 18;
  float acc = bias[co];
  const float* wr = w + co * 48; // [ci][3]
#pragma unroll
  for (int dl = 0; dl < 3; ++dl) {
    int ll = l + dl - 1;
    if ((unsigned)ll < 1024u) {
      const float* xr = x + ((size_t)(b * 1024 + ll)) * CIN;
#pragma unroll
      for (int ci = 0; ci < CIN; ++ci) acc += xr[ci] * wr[ci * 3 + dl];
    }
  }
  h1[gid] = (half_t)fmaxf(acc, 0.f);
}

// ---------------- conv2: 256->256 k=3 as 3 shifted WMMA GEMMs, bias+relu ----------------
__global__ void k_conv2(const half_t* __restrict__ H1, const half_t* __restrict__ W2t,
                        const float* __restrict__ bias, float* __restrict__ out) {
  int wid = (blockIdx.x * blockDim.x + threadIdx.x) >> 5;
  int lane = threadIdx.x & 31;
  if (wid >= (MROWS >> 4) * 16) return;
  int row0 = (wid >> 4) << 4, n0 = (wid & 15) << 4;
  int mlane = row0 + (lane & 15);
  int l = mlane & 1023;
  v8f c = {};
  for (int dl = 0; dl < 3; ++dl) {
    bool ok = (unsigned)(l + dl - 1) < 1024u; // batch-boundary zero pad
    const half_t* wt = W2t + dl * 65536;
#pragma unroll
    for (int k0 = 0; k0 < 256; k0 += 32) {
      v16h a = {};
      if (ok) a = load_a(H1, 256, mlane + dl - 1, k0, lane);
      v16h b = load_b(wt, 256, n0, k0, lane);
      c = wmma16(a, b, c);
    }
  }
  int n = n0 + (lane & 15);
  float bv = bias[n];
  int mbase = row0 + ((lane & 16) ? 8 : 0);
#pragma unroll
  for (int r = 0; r < 8; ++r)
    out[(size_t)(mbase + r) * DM + n] = fmaxf(c[r] + bv, 0.f);
}

// ---------------- layernorm over width 256, one wave per row ----------------
__global__ void k_ln(const float* __restrict__ X, const float* __restrict__ g,
                     const float* __restrict__ be, float* __restrict__ O32,
                     half_t* __restrict__ O16, int rows) {
  int wid = (blockIdx.x * blockDim.x + threadIdx.x) >> 5;
  int lane = threadIdx.x & 31;
  if (wid >= rows) return;
  const float* x = X + (size_t)wid * 256;
  float vals[8], s = 0.f;
#pragma unroll
  for (int i = 0; i < 8; ++i) { vals[i] = x[lane + i * 32]; s += vals[i]; }
  for (int o = 16; o; o >>= 1) s += __shfl_xor(s, o, 32);
  float mean = s * (1.f / 256.f), vs = 0.f;
#pragma unroll
  for (int i = 0; i < 8; ++i) { float d = vals[i] - mean; vs += d * d; }
  for (int o = 16; o; o >>= 1) vs += __shfl_xor(vs, o, 32);
  float inv = rsqrtf(vs * (1.f / 256.f) + 1e-5f);
#pragma unroll
  for (int i = 0; i < 8; ++i) {
    int col = lane + i * 32;
    float o = (vals[i] - mean) * inv * g[col] + be[col];
    if (O32) O32[(size_t)wid * 256 + col] = o;
    if (O16) O16[(size_t)wid * 256 + col] = (half_t)o;
  }
}

// ------- autocorrelation mean: circular-diagonal sums of per-batch Gram Q Kt -------
// One block owns (batch, 16-row Q strip). The 8 A fragments of the strip are
// pre-loaded into registers once; 8 waves sweep all 64 K-tiles; diagonal sums
// accumulate in LDS (ds_add_f32) and flush with 1024 global atomics per block.
__global__ void k_corrmean(const half_t* __restrict__ Q, const half_t* __restrict__ Kc,
                           float* __restrict__ meanv) {
  __shared__ float acc[LL];
  int b = blockIdx.x >> 6;
  int tp0 = (blockIdx.x & 63) << 4;
  int tid = threadIdx.x;
  int wv = tid >> 5, lane = tid & 31;
  for (int i = tid; i < LL; i += 256) acc[i] = 0.f;
  __syncthreads();
  const half_t* Qb = Q + (size_t)b * LL * DM;
  const half_t* Kb = Kc + (size_t)b * LL * DM; // row-major == Bt layout for K^T
  v16h af[8];
#pragma unroll
  for (int i = 0; i < 8; ++i) af[i] = load_a(Qb, 256, tp0 + (lane & 15), i * 32, lane);
  int tl = lane & 15;
  int mbase = tp0 + ((lane & 16) ? 8 : 0);
  for (int j = 0; j < 8; ++j) {
    int t0 = ((wv << 3) + j) << 4;
    v8f c = {};
#pragma unroll
    for (int i = 0; i < 8; ++i) {
      v16h bf = load_b(Kb, 256, t0, i * 32, lane);
      c = wmma16(af[i], bf, c);
    }
#pragma unroll
    for (int r = 0; r < 8; ++r) {
      int tau = (mbase + r - (t0 + tl)) & 1023;
      atomicAdd(&acc[tau], c[r] * (1.f / 256.f));
    }
  }
  __syncthreads();
  float* mv = meanv + b * LL;
  for (int i = tid; i < LL; i += 256) atomicAdd(mv + i, acc[i]);
}

// ---------------- top-20 + softmax, one wave per batch ----------------
__global__ void k_topk(const float* __restrict__ meanv, float* __restrict__ wts,
                       int* __restrict__ dly) {
  __shared__ float sm[LL];
  int b = blockIdx.x, lane = threadIdx.x;
  const float* src = meanv + b * LL;
  for (int i = lane; i < LL; i += 32) sm[i] = src[i];
  __syncthreads();
  float topv[KTOP]; int topi[KTOP];
  for (int j = 0; j < KTOP; ++j) {
    float best = -1e30f; int bi = 0;
    for (int i = lane; i < LL; i += 32) { float v = sm[i]; if (v > best) { best = v; bi = i; } }
    for (int o = 16; o; o >>= 1) {
      float ov = __shfl_xor(best, o, 32);
      int oi = __shfl_xor(bi, o, 32);
      if (ov > best) { best = ov; bi = oi; }
    }
    topv[j] = best; topi[j] = bi;
    if (lane == 0) sm[bi] = -1e30f;
    __syncthreads();
  }
  float mx = topv[0];
  for (int j = 1; j < KTOP; ++j) mx = fmaxf(mx, topv[j]);
  float se = 0.f;
  for (int j = 0; j < KTOP; ++j) se += expf(topv[j] - mx);
  if (lane < KTOP) {
    wts[b * KTOP + lane] = expf(topv[lane] - mx) / se;
    dly[b * KTOP + lane] = topi[lane];
  }
}

// ---------------- weighted circular roll-gather of V ----------------
__global__ void k_aggregate(const half_t* __restrict__ V, const float* __restrict__ wts,
                            const int* __restrict__ dly, half_t* __restrict__ AC) {
  __shared__ float sw[KTOP];
  __shared__ int   sd[KTOP];
  int gid = blockIdx.x * blockDim.x + threadIdx.x;
  int b = gid >> 18; // uniform within a 256-thread block
  if (threadIdx.x < KTOP) {
    sw[threadIdx.x] = wts[b * KTOP + threadIdx.x];
    sd[threadIdx.x] = dly[b * KTOP + threadIdx.x];
  }
  __syncthreads();
  if (gid >= MROWS * DM) return;
  int d = gid & 255, l = (gid >> 8) & 1023;
  float acc = 0.f;
#pragma unroll
  for (int j = 0; j < KTOP; ++j) {
    int idx = (l + sd[j]) & 1023;
    acc += sw[j] * (float)V[(((size_t)(b * 1024 + idx)) << 8) + d];
  }
  AC[gid] = (half_t)acc;
}

// ---------------- series_decomp: x - movavg(x,25) with replicate pad ----------------
__global__ void k_decomp(const float* __restrict__ X, float* __restrict__ S32,
                         half_t* __restrict__ S16) {
  int gid = blockIdx.x * blockDim.x + threadIdx.x;
  if (gid >= MROWS * DM) return;
  int d = gid & 255, l = (gid >> 8) & 1023, b = gid >> 18;
  float s = 0.f;
#pragma unroll
  for (int j = -12; j <= 12; ++j) {
    int ll = min(max(l + j, 0), 1023);
    s += X[(((size_t)(b * 1024 + ll)) << 8) + d];
  }
  float v = X[gid] - s * (1.f / 25.f);
  S32[gid] = v;
  S16[gid] = (half_t)v;
}

__global__ void k_pool(const float* __restrict__ enc, float* __restrict__ dst) {
  int gid = blockIdx.x * blockDim.x + threadIdx.x;
  if (gid >= 64 * 256) return;
  int b = gid >> 8, d = gid & 255;
  dst[gid] = enc[(((size_t)(b * 1024 + 1023)) << 8) + d];
}

__global__ void k_concat(const float* __restrict__ p1, const float* __restrict__ p2,
                         const float* __restrict__ emb, half_t* __restrict__ dst) {
  int gid = blockIdx.x * blockDim.x + threadIdx.x;
  if (gid >= 64 * 768) return;
  int c = gid % 768, b = gid / 768;
  float v = (c < 256) ? p1[b * 256 + c] : (c < 512) ? p2[b * 256 + c - 256]
                                                    : emb[b * 256 + c - 512];
  dst[gid] = (half_t)v;
}

__global__ void k_final(const float* __restrict__ R, const float* __restrict__ w,
                        const float* __restrict__ b2, float* __restrict__ out) {
  int b = blockIdx.x, lane = threadIdx.x;
  float s = 0.f;
  for (int i = lane; i < 256; i += 32) s += R[b * 256 + i] * w[i];
  for (int o = 16; o; o >>= 1) s += __shfl_xor(s, o, 32);
  if (lane == 0) out[b] = s + b2[0];
}

// =======================================================================
extern "C" void kernel_launch(void* const* d_in, const int* in_sizes, int n_in,
                              void* d_out, int out_size, void* d_ws, size_t ws_size,
                              hipStream_t stream) {
  const float* x_enc   = (const float*)d_in[0];
  const float* conv1_w = (const float*)d_in[1];
  const float* conv1_b = (const float*)d_in[2];
  const float* conv2_w = (const float*)d_in[3];
  const float* conv2_b = (const float*)d_in[4];
  const float* cnn_g   = (const float*)d_in[5];
  const float* cnn_be  = (const float*)d_in[6];
  const float* proj_w  = (const float*)d_in[7];
  const float* proj_b  = (const float*)d_in[8];
  const float* pln_g   = (const float*)d_in[9];
  const float* pln_b   = (const float*)d_in[10];
  const float* Wq = (const float*)d_in[11]; const float* bq = (const float*)d_in[12];
  const float* Wk = (const float*)d_in[13]; const float* bk = (const float*)d_in[14];
  const float* Wv = (const float*)d_in[15]; const float* bv = (const float*)d_in[16];
  const float* Wo = (const float*)d_in[17]; const float* bo = (const float*)d_in[18];
  const float* Wff1 = (const float*)d_in[19];
  const float* Wff2 = (const float*)d_in[20];
  const float* rp1_w = (const float*)d_in[21]; const float* rp1_b = (const float*)d_in[22];
  const float* rln_g = (const float*)d_in[23]; const float* rln_b = (const float*)d_in[24];
  const float* rp2_w = (const float*)d_in[25]; const float* rp2_b = (const float*)d_in[26];
  float* out = (float*)d_out;

  char* ws = (char*)d_ws;
  float*  ENC32 = (float*)(ws + OFF_ENC32);
  half_t* ENC16 = (half_t*)(ws + OFF_ENC16);
  float*  XS32  = ENC32;   // aliased: enc dead when xs live and vice versa
  half_t* XS16  = ENC16;
  half_t* Q16 = (half_t*)(ws + OFF_Q16);
  half_t* K16 = (half_t*)(ws + OFF_K16);
  half_t* V16 = (half_t*)(ws + OFF_V16);
  float*  X32 = (float*)(ws + OFF_X32);
  half_t* Y16 = (half_t*)(ws + OFF_Y16); // FFN hidden; also conv1 H1
  half_t* WP  = (half_t*)(ws + OFF_WP);
  char* SMp = ws + OFF_SM;
  float* meanv = (float*)(SMp + SM_MEANV);
  float* wts   = (float*)(SMp + SM_W);
  int*   dly   = (int*)(SMp + SM_D);
  float* pool1 = (float*)(SMp + SM_P1);
  float* pool2 = (float*)(SMp + SM_P2);
  float* epre  = (float*)(SMp + SM_EPRE);
  float* emb   = (float*)(SMp + SM_EMB);
  float* rpre  = (float*)(SMp + SM_RPRE);
  float* rpn   = (float*)(SMp + SM_RPN);

  auto pack = [&](const float* s, half_t* d, int N, int K, int sn, int sk, int off) {
    int tot = N * K;
    k_pack<<<(tot + 255) / 256, 256, 0, stream>>>(s, d, N, K, sn, sk, off);
  };
  auto gemm = [&](const half_t* A, const half_t* Bt, const float* bias,
                  const float* resid, float* o32, half_t* o16,
                  int M, int N, int K, int act) {
    int waves = (M / 32) * (N / 32);
    k_gemm<<<(waves * 32 + 255) / 256, 256, 0, stream>>>(A, Bt, bias, resid, o32, o16,
                                                         M, N, K, act);
  };

  // --- weight packing (f32 -> f16, B-transposed [N,K]) ---
  for (int i = 0; i < 2; ++i) {
    pack(Wq + i * 65536, WP + WP_WQ + i * 65536, 256, 256, 1, 256, 0);
    pack(Wk + i * 65536, WP + WP_WK + i * 65536, 256, 256, 1, 256, 0);
    pack(Wv + i * 65536, WP + WP_WV + i * 65536, 256, 256, 1, 256, 0);
    pack(Wo + i * 65536, WP + WP_WO + i * 65536, 256, 256, 1, 256, 0);
    pack(Wff1 + i * 262144, WP + WP_FF1 + i * 262144, 1024, 256, 256, 1, 0); // already [N,K]
    pack(Wff2 + i * 262144, WP + WP_FF2 + i * 262144, 256, 1024, 1024, 1, 0);
  }
  for (int dl = 0; dl < 3; ++dl)
    pack(conv2_w, WP + WP_C2 + dl * 65536, 256, 256, 768, 3, dl);
  pack(proj_w, WP + WP_PROJ, 256, 16384, 1, 256, 0);
  pack(rp1_w, WP + WP_RP1, 256, 768, 1, 256, 0);
  pack(x_enc, WP + WP_XE, 64, 16384, 16384, 1, 0); // flattened x_enc in f16

  // --- CNN frontend ---
  k_conv1<<<(MROWS * DM) / 256, 256, 0, stream>>>(x_enc, conv1_w, conv1_b, Y16);
  {
    int waves = (MROWS / 16) * 16;
    k_conv2<<<(waves * 32) / 256, 256, 0, stream>>>(Y16, WP + WP_C2, conv2_b, X32);
  }
  k_ln<<<(MROWS * 32) / 256, 256, 0, stream>>>(X32, cnn_g, cnn_be, ENC32, ENC16, MROWS);

  // --- input embedding branch ---
  gemm(WP + WP_XE, WP + WP_PROJ, proj_b, nullptr, epre, nullptr, 64, 256, 16384, 1);
  k_ln<<<(64 * 32) / 256, 256, 0, stream>>>(epre, pln_g, pln_b, emb, nullptr, 64);

  // --- encoder layers ---
  for (int i = 0; i < 2; ++i) {
    gemm(ENC16, WP + WP_WQ + i * 65536, bq + i * 256, nullptr, nullptr, Q16, MROWS, 256, 256, 0);
    gemm(ENC16, WP + WP_WK + i * 65536, bk + i * 256, nullptr, nullptr, K16, MROWS, 256, 256, 0);
    gemm(ENC16, WP + WP_WV + i * 65536, bv + i * 256, nullptr, nullptr, V16, MROWS, 256, 256, 0);
    k_zero<<<(BB * LL) / 256, 256, 0, stream>>>(meanv, BB * LL);
    k_corrmean<<<BB * 64, 256, 0, stream>>>(Q16, K16, meanv);
    k_topk<<<BB, 32, 0, stream>>>(meanv, wts, dly);
    k_aggregate<<<(MROWS * DM) / 256, 256, 0, stream>>>(V16, wts, dly, Q16); // AC -> Q16
    gemm(Q16, WP + WP_WO + i * 65536, bo + i * 256, ENC32, X32, nullptr, MROWS, 256, 256, 0);
    k_decomp<<<(MROWS * DM) / 256, 256, 0, stream>>>(X32, XS32, XS16);
    gemm(XS16, WP + WP_FF1 + i * 262144, nullptr, nullptr, nullptr, Y16, MROWS, 1024, 256, 2);
    gemm(Y16, WP + WP_FF2 + i * 262144, nullptr, XS32, X32, nullptr, MROWS, 256, 1024, 0);
    k_decomp<<<(MROWS * DM) / 256, 256, 0, stream>>>(X32, ENC32, ENC16);
    k_pool<<<(64 * 256) / 256, 256, 0, stream>>>(ENC32, i == 0 ? pool1 : pool2);
  }

  // --- regression head ---
  k_concat<<<(64 * 768) / 256, 256, 0, stream>>>(pool1, pool2, emb, WP + WP_CAT);
  gemm(WP + WP_CAT, WP + WP_RP1, rp1_b, nullptr, rpre, nullptr, 64, 256, 768, 1);
  k_ln<<<(64 * 32) / 256, 256, 0, stream>>>(rpre, rln_g, rln_b, rpn, nullptr, 64);
  k_final<<<64, 32, 0, stream>>>(rpn, rp2_w, rp2_b, out);
}
